// EPSparseMoeBlock_12300786335900
// MI455X (gfx1250) — matrix operand order repaired and checked by reading, not compiled, not run on hardware
//
#include <hip/hip_runtime.h>
#include <hip/hip_bf16.h>
#include <math.h>
#include <stdint.h>

// ---------------- problem constants ----------------
constexpr int TT = 4096;          // tokens = B*L
constexpr int DD = 2048;          // hidden dim
constexpr int EE = 8;             // experts
constexpr int II = 768;           // expert intermediate
constexpr int IS = 4096;          // shared intermediate
constexpr int BM = 128;           // GEMM tile M
constexpr int BN = 128;           // GEMM tile N
constexpr int BK = 32;            // GEMM tile K (bf16 wmma K)
constexpr int LDSK = 40;          // padded LDS row stride (elems) = 80B
constexpr int SLOT_CAP = TT * 2 + EE * BM;   // 9216 padded slots

typedef __bf16 bf16_t;
typedef __attribute__((ext_vector_type(8)))  __bf16 v8bf;
typedef __attribute__((ext_vector_type(16))) __bf16 v16bf;
typedef __attribute__((ext_vector_type(8)))  float  v8f;
typedef __attribute__((ext_vector_type(4)))  float  v4f;
typedef int v4i_t __attribute__((vector_size(16)));

// ---------------- CDNA5 async global->LDS copy (16B per lane) ----------------
#if defined(__AMDGCN__) && __has_builtin(__builtin_amdgcn_global_load_async_to_lds_b128)
#define ASYNC_COPY 1
#else
#define ASYNC_COPY 0
#endif

__device__ __forceinline__ void copy16(bf16_t* l, const bf16_t* g) {
#if ASYNC_COPY
  __builtin_amdgcn_global_load_async_to_lds_b128(
      (__attribute__((address_space(1))) v4i_t*)(uintptr_t)g,
      (__attribute__((address_space(3))) v4i_t*)(unsigned int)(uintptr_t)l,
      0, 0);
#else
  *(v8bf*)l = *(const v8bf*)g;
#endif
}

__device__ __forceinline__ void wait_async() {
#if ASYNC_COPY
#if __has_builtin(__builtin_amdgcn_s_wait_asynccnt)
  __builtin_amdgcn_s_wait_asynccnt(0);
#else
  asm volatile("s_wait_asynccnt 0x0" ::: "memory");
#endif
#endif
}

// fast sigmoid: v_rcp_f32 instead of full IEEE divide sequence
__device__ __forceinline__ float fast_sigmoid(float x) {
  return __builtin_amdgcn_rcpf(1.f + __expf(-x));
}

// ---------------- fp32 -> bf16 conversion ----------------
__global__ __launch_bounds__(256) void cvt_kernel(const float* __restrict__ s,
                                                  bf16_t* __restrict__ d,
                                                  long long n4) {
  long long i = blockIdx.x * (long long)blockDim.x + threadIdx.x;
  long long stride = gridDim.x * (long long)blockDim.x;
  for (; i < n4; i += stride) {
    v4f f = ((const v4f*)s)[i];
    union { bf16_t b[4]; unsigned long long q; } o;
    o.b[0] = (bf16_t)f.x; o.b[1] = (bf16_t)f.y;
    o.b[2] = (bf16_t)f.z; o.b[3] = (bf16_t)f.w;
    ((unsigned long long*)d)[i] = o.q;
  }
}

// ---------------- router: softmax -> top2 -> renorm, + shared scalar gate ----------------
__global__ __launch_bounds__(256) void router_kernel(const float* __restrict__ h,
                                                     const float* __restrict__ gate_w,
                                                     const float* __restrict__ seg_w,
                                                     int* __restrict__ topk_idx,
                                                     float* __restrict__ topk_w,
                                                     float* __restrict__ sgate,
                                                     int* __restrict__ counts) {
  int wave = threadIdx.x >> 5;
  int lane = threadIdx.x & 31;
  int t = blockIdx.x * 8 + wave;
  if (t >= TT) return;
  const float* hr = h + (long long)t * DD;
  float acc[9];
#pragma unroll
  for (int e = 0; e < 9; ++e) acc[e] = 0.f;
  for (int dd = lane; dd < DD; dd += 32) {
    float hv = hr[dd];
#pragma unroll
    for (int e = 0; e < 8; ++e) acc[e] += hv * gate_w[e * DD + dd];
    acc[8] += hv * seg_w[dd];
  }
#pragma unroll
  for (int off = 16; off >= 1; off >>= 1) {
#pragma unroll
    for (int e = 0; e < 9; ++e) acc[e] += __shfl_xor(acc[e], off, 32);
  }
  if (lane == 0) {
    float m = acc[0];
#pragma unroll
    for (int e = 1; e < 8; ++e) m = fmaxf(m, acc[e]);
    float p[8], ps = 0.f;
#pragma unroll
    for (int e = 0; e < 8; ++e) { p[e] = __expf(acc[e] - m); ps += p[e]; }
    float rps = __builtin_amdgcn_rcpf(ps);
#pragma unroll
    for (int e = 0; e < 8; ++e) p[e] *= rps;
    int i1 = 0;
#pragma unroll
    for (int e = 1; e < 8; ++e) if (p[e] > p[i1]) i1 = e;
    int i2 = (i1 == 0) ? 1 : 0;
#pragma unroll
    for (int e = 0; e < 8; ++e) if (e != i1 && p[e] > p[i2]) i2 = e;
    float rs2 = __builtin_amdgcn_rcpf(p[i1] + p[i2]);
    topk_idx[t * 2 + 0] = i1; topk_idx[t * 2 + 1] = i2;
    topk_w[t * 2 + 0] = p[i1] * rs2; topk_w[t * 2 + 1] = p[i2] * rs2;
    sgate[t] = fast_sigmoid(acc[8]);
    atomicAdd(&counts[i1], 1);
    atomicAdd(&counts[i2], 1);
  }
}

// ---------------- 128-aligned per-expert slot offsets ----------------
__global__ void offsets_kernel(const int* __restrict__ counts, int* __restrict__ offs) {
  if (threadIdx.x == 0 && blockIdx.x == 0) {
    int o = 0;
    for (int e = 0; e < EE; ++e) {
      offs[e] = o;
      o += ((counts[e] + BM - 1) / BM) * BM;
    }
    offs[EE] = o;
  }
}

// ---------------- scatter tokens to compacted expert slots ----------------
__global__ __launch_bounds__(256) void scatter_kernel(const int* __restrict__ topk_idx,
                                                      const int* __restrict__ offs,
                                                      int* __restrict__ cursors,
                                                      int* __restrict__ slot_token,
                                                      int* __restrict__ slot_of) {
  int t = blockIdx.x * blockDim.x + threadIdx.x;
  if (t >= TT) return;
#pragma unroll
  for (int k = 0; k < 2; ++k) {
    int e = topk_idx[t * 2 + k];
    int pos = offs[e] + atomicAdd(&cursors[e], 1);
    slot_token[pos] = t;
    slot_of[t * 2 + k] = pos;
  }
}

// ---------------- WMMA fragment helpers (bf16, wave32) ----------------
// A 16x32 bf16 frag: lane (m + 16h) holds K {8h..8h+7} and {16+8h..16+8h+7} of row m.
__device__ __forceinline__ v16bf frag_a(const bf16_t* lds, int row0) {
  int lane = threadIdx.x & 31;
  const bf16_t* p = lds + (row0 + (lane & 15)) * LDSK + ((lane >> 4) << 3);
  union { v16bf v; v8bf h[2]; } u;
  u.h[0] = *(const v8bf*)p;
  u.h[1] = *(const v8bf*)(p + 16);
  return u.v;
}
// B 32x16 bf16 frag from [N][K] storage: lane (n + 16h) holds K {8h..8h+7} and {16+8h..16+8h+7} of col n.
__device__ __forceinline__ v16bf frag_b(const bf16_t* lds, int col0) {
  int lane = threadIdx.x & 31;
  const bf16_t* p = lds + (col0 + (lane & 15)) * LDSK + ((lane >> 4) << 3);
  union { v16bf v; v8bf h[2]; } u;
  u.h[0] = *(const v8bf*)p;
  u.h[1] = *(const v8bf*)(p + 16);
  return u.v;
}
__device__ __forceinline__ v8f wmma_bf16(v16bf a, v16bf b, v8f c) {
  return __builtin_amdgcn_wmma_f32_16x16x32_bf16(false, a, false, b, (short)0, c, false, false);
}

// ---------------- shared expert GEMM1: act = silu(H Wg^T) * (H Wu^T) ----------------
__global__ __launch_bounds__(256) void shared_gemm1(const bf16_t* __restrict__ hbf,
                                                    const bf16_t* __restrict__ wsg,
                                                    const bf16_t* __restrict__ wsu,
                                                    bf16_t* __restrict__ act) {
  __shared__ alignas(16) bf16_t lA[2][BM * LDSK];
  __shared__ alignas(16) bf16_t lBg[2][BN * LDSK];
  __shared__ alignas(16) bf16_t lBu[2][BN * LDSK];
  int mt = blockIdx.x, nt = blockIdx.y;
  int tid = threadIdx.x;
  int cr = tid >> 1, cc = (tid & 1) << 4;
  const bf16_t* gA  = hbf + (long long)(mt * BM + cr) * DD + cc;
  const bf16_t* gBg = wsg + (long long)(nt * BN + cr) * DD + cc;
  const bf16_t* gBu = wsu + (long long)(nt * BN + cr) * DD + cc;
  bf16_t* wA[2]  = { &lA[0][cr * LDSK + cc],  &lA[1][cr * LDSK + cc] };
  bf16_t* wBg[2] = { &lBg[0][cr * LDSK + cc], &lBg[1][cr * LDSK + cc] };
  bf16_t* wBu[2] = { &lBu[0][cr * LDSK + cc], &lBu[1][cr * LDSK + cc] };
  auto issue = [&](int b, int k0) {
    copy16(wA[b],      gA + k0);  copy16(wA[b] + 8,  gA + k0 + 8);
    copy16(wBg[b],     gBg + k0); copy16(wBg[b] + 8, gBg + k0 + 8);
    copy16(wBu[b],     gBu + k0); copy16(wBu[b] + 8, gBu + k0 + 8);
  };
  int wave = tid >> 5;
  int wrow = (wave >> 2) * 64, wcol = (wave & 3) * 32;
  v8f zero = {0.f, 0.f, 0.f, 0.f, 0.f, 0.f, 0.f, 0.f};
  v8f accg[4][2], accu[4][2];
#pragma unroll
  for (int i = 0; i < 4; ++i)
#pragma unroll
    for (int j = 0; j < 2; ++j) { accg[i][j] = zero; accu[i][j] = zero; }

  constexpr int ITERS = DD / BK;
  issue(0, 0);
  for (int it = 0; it < ITERS; ++it) {
    wait_async();
    __syncthreads();
    if (it + 1 < ITERS) issue((it + 1) & 1, (it + 1) * BK);
    const bf16_t* bA  = lA[it & 1];
    const bf16_t* bBg = lBg[it & 1];
    const bf16_t* bBu = lBu[it & 1];
    v16bf af[4];
#pragma unroll
    for (int tm = 0; tm < 4; ++tm) af[tm] = frag_a(bA, wrow + tm * 16);
#pragma unroll
    for (int tn = 0; tn < 2; ++tn) {
      v16bf bg = frag_b(bBg, wcol + tn * 16);
      v16bf bu = frag_b(bBu, wcol + tn * 16);
#pragma unroll
      for (int tm = 0; tm < 4; ++tm) {
        accg[tm][tn] = wmma_bf16(af[tm], bg, accg[tm][tn]);
        accu[tm][tn] = wmma_bf16(af[tm], bu, accu[tm][tn]);
      }
    }
  }
  int lane = tid & 31, hh = lane >> 4, nl = lane & 15;
#pragma unroll
  for (int tm = 0; tm < 4; ++tm)
#pragma unroll
    for (int tn = 0; tn < 2; ++tn)
#pragma unroll
      for (int r = 0; r < 8; ++r) {
        int m = mt * BM + wrow + tm * 16 + hh * 8 + r;
        int n = nt * BN + wcol + tn * 16 + nl;
        float g = accg[tm][tn][r], u = accu[tm][tn][r];
        act[(long long)m * IS + n] = (bf16_t)(g * fast_sigmoid(g) * u);
      }
}

// ---------------- shared expert GEMM2: out = sigmoid_gate * (act Wd^T) ----------------
__global__ __launch_bounds__(256) void shared_gemm2(const bf16_t* __restrict__ act,
                                                    const bf16_t* __restrict__ wsd,
                                                    const float* __restrict__ sgate,
                                                    float* __restrict__ out) {
  __shared__ alignas(16) bf16_t lA[2][BM * LDSK];
  __shared__ alignas(16) bf16_t lB[2][BN * LDSK];
  int mt = blockIdx.x, nt = blockIdx.y;
  int tid = threadIdx.x;
  int cr = tid >> 1, cc = (tid & 1) << 4;
  const bf16_t* gA = act + (long long)(mt * BM + cr) * IS + cc;
  const bf16_t* gB = wsd + (long long)(nt * BN + cr) * IS + cc;
  bf16_t* wAp[2] = { &lA[0][cr * LDSK + cc], &lA[1][cr * LDSK + cc] };
  bf16_t* wBp[2] = { &lB[0][cr * LDSK + cc], &lB[1][cr * LDSK + cc] };
  auto issue = [&](int b, int k0) {
    copy16(wAp[b],     gA + k0); copy16(wAp[b] + 8, gA + k0 + 8);
    copy16(wBp[b],     gB + k0); copy16(wBp[b] + 8, gB + k0 + 8);
  };
  int wave = tid >> 5;
  int wrow = (wave >> 2) * 64, wcol = (wave & 3) * 32;
  v8f zero = {0.f, 0.f, 0.f, 0.f, 0.f, 0.f, 0.f, 0.f};
  v8f acc[4][2];
#pragma unroll
  for (int i = 0; i < 4; ++i)
#pragma unroll
    for (int j = 0; j < 2; ++j) acc[i][j] = zero;

  constexpr int ITERS = IS / BK;
  issue(0, 0);
  for (int it = 0; it < ITERS; ++it) {
    wait_async();
    __syncthreads();
    if (it + 1 < ITERS) issue((it + 1) & 1, (it + 1) * BK);
    const bf16_t* bA = lA[it & 1];
    const bf16_t* bB = lB[it & 1];
    v16bf af[4];
#pragma unroll
    for (int tm = 0; tm < 4; ++tm) af[tm] = frag_a(bA, wrow + tm * 16);
#pragma unroll
    for (int tn = 0; tn < 2; ++tn) {
      v16bf b = frag_b(bB, wcol + tn * 16);
#pragma unroll
      for (int tm = 0; tm < 4; ++tm) acc[tm][tn] = wmma_bf16(af[tm], b, acc[tm][tn]);
    }
  }
  int lane = tid & 31, hh = lane >> 4, nl = lane & 15;
#pragma unroll
  for (int tm = 0; tm < 4; ++tm)
#pragma unroll
    for (int tn = 0; tn < 2; ++tn)
#pragma unroll
      for (int r = 0; r < 8; ++r) {
        int m = mt * BM + wrow + tm * 16 + hh * 8 + r;
        int n = nt * BN + wcol + tn * 16 + nl;
        out[(long long)m * DD + n] = sgate[m] * acc[tm][tn][r];
      }
}

// ---------------- MoE GEMM1 (gathered rows): act = silu(g)*u, per-expert ----------------
__global__ __launch_bounds__(256) void moe_gemm1(const bf16_t* __restrict__ hbf,
                                                 const bf16_t* __restrict__ wgu,
                                                 const int* __restrict__ counts,
                                                 const int* __restrict__ offs,
                                                 const int* __restrict__ slot_token,
                                                 bf16_t* __restrict__ act) {
  int e = blockIdx.x >> 5;
  int mt = blockIdx.x & 31;
  int nt = blockIdx.y;
  int cnt = counts[e];
  if (mt * BM >= cnt) return;
  int slot_base = offs[e] + mt * BM;
  int rows_valid = min(BM, cnt - mt * BM);

  __shared__ alignas(16) bf16_t lA[2][BM * LDSK];
  __shared__ alignas(16) bf16_t lBg[2][BN * LDSK];
  __shared__ alignas(16) bf16_t lBu[2][BN * LDSK];
  int tid = threadIdx.x;
  int cr = tid >> 1, cc = (tid & 1) << 4;
  // pad rows clamp to slot 0: their outputs land in pad rows never read downstream
  int rr = (cr < rows_valid) ? cr : 0;
  int tok = slot_token[slot_base + rr];
  const bf16_t* gA  = hbf + (long long)tok * DD + cc;
  const bf16_t* Bgs = wgu + ((long long)e * 2 * II + (long long)nt * BN) * DD;
  const bf16_t* gBg = Bgs + (long long)cr * DD + cc;
  const bf16_t* gBu = gBg + (long long)II * DD;
  bf16_t* wA[2]  = { &lA[0][cr * LDSK + cc],  &lA[1][cr * LDSK + cc] };
  bf16_t* wBg[2] = { &lBg[0][cr * LDSK + cc], &lBg[1][cr * LDSK + cc] };
  bf16_t* wBu[2] = { &lBu[0][cr * LDSK + cc], &lBu[1][cr * LDSK + cc] };
  auto issue = [&](int b, int k0) {
    copy16(wA[b],      gA + k0);  copy16(wA[b] + 8,  gA + k0 + 8);
    copy16(wBg[b],     gBg + k0); copy16(wBg[b] + 8, gBg + k0 + 8);
    copy16(wBu[b],     gBu + k0); copy16(wBu[b] + 8, gBu + k0 + 8);
  };
  int wave = tid >> 5;
  int wrow = (wave >> 2) * 64, wcol = (wave & 3) * 32;
  v8f zero = {0.f, 0.f, 0.f, 0.f, 0.f, 0.f, 0.f, 0.f};
  v8f accg[4][2], accu[4][2];
#pragma unroll
  for (int i = 0; i < 4; ++i)
#pragma unroll
    for (int j = 0; j < 2; ++j) { accg[i][j] = zero; accu[i][j] = zero; }

  constexpr int ITERS = DD / BK;
  issue(0, 0);
  for (int it = 0; it < ITERS; ++it) {
    wait_async();
    __syncthreads();
    if (it + 1 < ITERS) issue((it + 1) & 1, (it + 1) * BK);
    const bf16_t* bA  = lA[it & 1];
    const bf16_t* bBg = lBg[it & 1];
    const bf16_t* bBu = lBu[it & 1];
    v16bf af[4];
#pragma unroll
    for (int tm = 0; tm < 4; ++tm) af[tm] = frag_a(bA, wrow + tm * 16);
#pragma unroll
    for (int tn = 0; tn < 2; ++tn) {
      v16bf bg = frag_b(bBg, wcol + tn * 16);
      v16bf bu = frag_b(bBu, wcol + tn * 16);
#pragma unroll
      for (int tm = 0; tm < 4; ++tm) {
        accg[tm][tn] = wmma_bf16(af[tm], bg, accg[tm][tn]);
        accu[tm][tn] = wmma_bf16(af[tm], bu, accu[tm][tn]);
      }
    }
  }
  int lane = tid & 31, hh = lane >> 4, nl = lane & 15;
#pragma unroll
  for (int tm = 0; tm < 4; ++tm)
#pragma unroll
    for (int tn = 0; tn < 2; ++tn)
#pragma unroll
      for (int r = 0; r < 8; ++r) {
        int ml = wrow + tm * 16 + hh * 8 + r;
        int n = nt * BN + wcol + tn * 16 + nl;
        float g = accg[tm][tn][r], u = accu[tm][tn][r];
        act[(long long)(slot_base + ml) * II + n] = (bf16_t)(g * fast_sigmoid(g) * u);
      }
}

// ---------------- MoE GEMM2: per-slot down-projection rows (fp32) ----------------
__global__ __launch_bounds__(256) void moe_gemm2(const bf16_t* __restrict__ act,
                                                 const bf16_t* __restrict__ wdn,
                                                 const int* __restrict__ counts,
                                                 const int* __restrict__ offs,
                                                 float* __restrict__ buf) {
  int e = blockIdx.x >> 5;
  int mt = blockIdx.x & 31;
  int nt = blockIdx.y;
  int cnt = counts[e];
  if (mt * BM >= cnt) return;
  int slot_base = offs[e] + mt * BM;

  __shared__ alignas(16) bf16_t lA[2][BM * LDSK];
  __shared__ alignas(16) bf16_t lB[2][BN * LDSK];
  int tid = threadIdx.x;
  int cr = tid >> 1, cc = (tid & 1) << 4;
  const bf16_t* gA = act + (long long)(slot_base + cr) * II + cc;
  const bf16_t* gB = wdn + (long long)e * DD * II + (long long)(nt * BN + cr) * II + cc;
  bf16_t* wAp[2] = { &lA[0][cr * LDSK + cc], &lA[1][cr * LDSK + cc] };
  bf16_t* wBp[2] = { &lB[0][cr * LDSK + cc], &lB[1][cr * LDSK + cc] };
  auto issue = [&](int b, int k0) {
    copy16(wAp[b],     gA + k0); copy16(wAp[b] + 8, gA + k0 + 8);
    copy16(wBp[b],     gB + k0); copy16(wBp[b] + 8, gB + k0 + 8);
  };
  int wave = tid >> 5;
  int wrow = (wave >> 2) * 64, wcol = (wave & 3) * 32;
  v8f zero = {0.f, 0.f, 0.f, 0.f, 0.f, 0.f, 0.f, 0.f};
  v8f acc[4][2];
#pragma unroll
  for (int i = 0; i < 4; ++i)
#pragma unroll
    for (int j = 0; j < 2; ++j) acc[i][j] = zero;

  constexpr int ITERS = II / BK;
  issue(0, 0);
  for (int it = 0; it < ITERS; ++it) {
    wait_async();
    __syncthreads();
    if (it + 1 < ITERS) issue((it + 1) & 1, (it + 1) * BK);
    const bf16_t* bA = lA[it & 1];
    const bf16_t* bB = lB[it & 1];
    v16bf af[4];
#pragma unroll
    for (int tm = 0; tm < 4; ++tm) af[tm] = frag_a(bA, wrow + tm * 16);
#pragma unroll
    for (int tn = 0; tn < 2; ++tn) {
      v16bf b = frag_b(bB, wcol + tn * 16);
#pragma unroll
      for (int tm = 0; tm < 4; ++tm) acc[tm][tn] = wmma_bf16(af[tm], b, acc[tm][tn]);
    }
  }
  int lane = tid & 31, hh = lane >> 4, nl = lane & 15;
#pragma unroll
  for (int tm = 0; tm < 4; ++tm)
#pragma unroll
    for (int tn = 0; tn < 2; ++tn)
#pragma unroll
      for (int r = 0; r < 8; ++r) {
        int ml = wrow + tm * 16 + hh * 8 + r;
        int n = nt * BN + wcol + tn * 16 + nl;
        buf[(long long)(slot_base + ml) * DD + n] = acc[tm][tn][r];
      }
}

// ---------------- final deterministic gather combine ----------------
__global__ __launch_bounds__(256) void combine_kernel(float* __restrict__ out,
                                                      const float* __restrict__ buf,
                                                      const int* __restrict__ slot_of,
                                                      const float* __restrict__ topk_w) {
  long long i = (blockIdx.x * (long long)blockDim.x + threadIdx.x) * 4;
  int t = (int)(i >> 11);           // DD == 2048
  int dcol = (int)(i & (DD - 1));
  int s0 = slot_of[t * 2 + 0], s1 = slot_of[t * 2 + 1];
  float w0 = topk_w[t * 2 + 0], w1 = topk_w[t * 2 + 1];
  v4f o  = *(const v4f*)(out + i);
  v4f b0 = *(const v4f*)(buf + (long long)s0 * DD + dcol);
  v4f b1 = *(const v4f*)(buf + (long long)s1 * DD + dcol);
  o = o + b0 * w0 + b1 * w1;
  *(v4f*)(out + i) = o;
}

// ---------------- host orchestration ----------------
extern "C" void kernel_launch(void* const* d_in, const int* in_sizes, int n_in,
                              void* d_out, int out_size, void* d_ws, size_t ws_size,
                              hipStream_t stream) {
  const float* h    = (const float*)d_in[0];
  const float* gw   = (const float*)d_in[1];
  const float* gup  = (const float*)d_in[2];
  const float* dnp  = (const float*)d_in[3];
  const float* sgw  = (const float*)d_in[4];
  const float* suw  = (const float*)d_in[5];
  const float* sdw  = (const float*)d_in[6];
  const float* segw = (const float*)d_in[7];
  float* out = (float*)d_out;

  char* ws = (char*)d_ws;
  size_t off = 0;
  auto alloc = [&](size_t bytes) -> char* {
    char* p = ws + off;
    off = (off + bytes + 255) & ~(size_t)255;
    return p;
  };
  bf16_t* hbf    = (bf16_t*)alloc((size_t)TT * DD * 2);
  bf16_t* wsg    = (bf16_t*)alloc((size_t)IS * DD * 2);
  bf16_t* wsu    = (bf16_t*)alloc((size_t)IS * DD * 2);
  bf16_t* wsd    = (bf16_t*)alloc((size_t)DD * IS * 2);
  bf16_t* wgu    = (bf16_t*)alloc((size_t)EE * 2 * II * DD * 2);
  bf16_t* wdn    = (bf16_t*)alloc((size_t)EE * DD * II * 2);
  bf16_t* act_sh = (bf16_t*)alloc((size_t)TT * IS * 2);
  bf16_t* act_mo = (bf16_t*)alloc((size_t)SLOT_CAP * II * 2);
  float*  buf    = (float*)alloc((size_t)SLOT_CAP * DD * 4);
  float*  sgate  = (float*)alloc((size_t)TT * 4);
  int*    tk_idx = (int*)alloc((size_t)TT * 2 * 4);
  float*  tk_w   = (float*)alloc((size_t)TT * 2 * 4);
  int*    sl_of  = (int*)alloc((size_t)TT * 2 * 4);
  int*    sl_tok = (int*)alloc((size_t)SLOT_CAP * 4);
  int*    counts = (int*)alloc(16 * 4);          // counts[8] + cursors[8] (one memset)
  int*    cursors = counts + 8;
  int*    offsets = (int*)alloc((EE + 1) * 4);

  (void)hipMemsetAsync(counts, 0, 16 * sizeof(int), stream);

  // fp32 -> bf16 conversions
  auto cvt = [&](const float* s, bf16_t* d, long long n) {
    long long n4 = n / 4;
    int grid = (int)((n4 + 255) / 256);
    if (grid > 16384) grid = 16384;
    cvt_kernel<<<grid, 256, 0, stream>>>(s, d, n4);
  };
  cvt(h,   hbf, (long long)TT * DD);
  cvt(sgw, wsg, (long long)IS * DD);
  cvt(suw, wsu, (long long)IS * DD);
  cvt(sdw, wsd, (long long)DD * IS);
  cvt(gup, wgu, (long long)EE * 2 * II * DD);
  cvt(dnp, wdn, (long long)EE * DD * II);

  // routing
  router_kernel<<<TT / 8, 256, 0, stream>>>(h, gw, segw, tk_idx, tk_w, sgate, counts);
  offsets_kernel<<<1, 32, 0, stream>>>(counts, offsets);
  scatter_kernel<<<TT / 256, 256, 0, stream>>>(tk_idx, offsets, cursors, sl_tok, sl_of);

  // shared expert (writes d_out)
  shared_gemm1<<<dim3(TT / BM, IS / BN), 256, 0, stream>>>(hbf, wsg, wsu, act_sh);
  shared_gemm2<<<dim3(TT / BM, DD / BN), 256, 0, stream>>>(act_sh, wsd, sgate, out);

  // sparse MoE
  moe_gemm1<<<dim3(EE * (TT / BM), II / BN), 256, 0, stream>>>(hbf, wgu, counts, offsets, sl_tok, act_mo);
  moe_gemm2<<<dim3(EE * (TT / BM), DD / BN), 256, 0, stream>>>(act_mo, wdn, counts, offsets, buf);

  // combine (adds weighted expert rows into d_out)
  combine_kernel<<<(int)(((long long)TT * DD / 4) / 256), 256, 0, stream>>>(out, buf, sl_of, tk_w);
}
